// TokenMixByCharBMM_67705864454577
// MI455X (gfx1250) — matrix operand order repaired and checked by reading, not compiled, not run on hardware
//
#include <hip/hip_runtime.h>
#include <hip/hip_bf16.h>

// ---------------------------------------------------------------------------
// TokenMixByCharBMM for MI455X (gfx1250, wave32, WMMA)
// Dominant cost: three [.,2048]x[2048,2048] GEMMs (~146 GFLOP) -> bf16 WMMA.
// Arithmetic intensity ~1000 FLOP/B -> compute bound; whole working set
// (<150 MB) is L2-resident (192 MB), so register-blocked WMMA with direct
// global loads + global_prefetch is the right data path.
// RoPE cancels exactly (same per-pair rotation applied to q and to every key
// of token t), so it is skipped. SWA zero-padding contributes score=0 softmax
// entries and v=0, matching the reference in exact arithmetic.
// ---------------------------------------------------------------------------

typedef __bf16 bf16;
typedef __attribute__((ext_vector_type(4)))  __bf16 v4bf;
typedef __attribute__((ext_vector_type(8)))  __bf16 v8bf;
typedef __attribute__((ext_vector_type(16))) __bf16 v16bf;
typedef __attribute__((ext_vector_type(8)))  float  v8f;

#define T_TOK 1024
#define CVN   8
#define DIM   2048
#define NHEAD 32
#define HDIM  64
#define WIN   8

// -------------------------------- f32 -> bf16 ------------------------------
__global__ __launch_bounds__(256) void cvt_f32_bf16(const float* __restrict__ s,
                                                    bf16* __restrict__ d, int n4) {
  int i = blockIdx.x * blockDim.x + threadIdx.x;
  if (i < n4) {
    float4 f = ((const float4*)s)[i];
    v4bf o;
    o[0] = (bf16)f.x; o[1] = (bf16)f.y; o[2] = (bf16)f.z; o[3] = (bf16)f.w;
    ((v4bf*)d)[i] = o;
  }
}

// ------------------------- bf16 WMMA GEMM: Y = X @ W^T ---------------------
// X: [M,K] bf16 row-major.  W: [N,K] bf16 row-major.  Y: [M,N] f32 or bf16.
// Block = 256 threads = 8 waves tiled 2(M) x 4(N); each wave owns a 64x64
// output tile = 4x4 v_wmma_f32_16x16x32_bf16 accumulators. Per 32-K step:
// one deep clause of 16 b128 loads (4 A frags + 4 B frags), then 16 WMMAs
// with staged s_wait_loadcnt so math starts as soon as operands land.
// Fragment layouts per CDNA5 ISA 7.12.2:
//   A (16x32 bf16): lanes 0-15 -> M=lane, elems 0..7 = K+0..7, 8..15 = K+16..23
//                   lanes 16-31 -> M=lane-16, K+8..15 and K+24..31
//   B (32x16 bf16): lane n (0-15) -> col n, elems 0..7 = K+0..7, 8..15 = K+16..23
//                   lane n+16     -> col n, K+8..15 and K+24..31
//   C/D (16x16 f32): lane half h, row lrow: elem r -> (M = 8h + r, N = lrow)
template <int STORE_BF16>
__global__ __launch_bounds__(256) void gemm_bf16(const bf16* __restrict__ X,
                                                 const bf16* __restrict__ W,
                                                 void* __restrict__ Y,
                                                 int M, int N, int K) {
  const int lane  = threadIdx.x & 31;
  const int wave  = threadIdx.x >> 5;
  const int half  = lane >> 4;
  const int lrow  = lane & 15;
  const int waveM = wave >> 2;  // 0..1
  const int waveN = wave & 3;   // 0..3

  const int tileM = blockIdx.x * 128 + waveM * 64;
  const int tileN = blockIdx.y * 256 + waveN * 64;

  v8f acc[4][4];
#pragma unroll
  for (int mi = 0; mi < 4; ++mi)
#pragma unroll
    for (int ni = 0; ni < 4; ++ni)
#pragma unroll
      for (int e = 0; e < 8; ++e) acc[mi][ni][e] = 0.0f;

  // Per-lane base pointers (row index is lane-dependent, K advances in loop).
  const bf16* aBase = X + (size_t)(tileM + lrow) * K + half * 8;   // + mi*16*K + kb
  const bf16* bBase = W + (size_t)(tileN + lrow) * K + half * 8;   // + ni*16*K + kb

  for (int kb = 0; kb < K; kb += 32) {
    // Prefetch the streams 8 K-steps ahead (gfx1250 global_prefetch_b8;
    // speculative, no LOADcnt, OOB silently dropped).
    __builtin_prefetch(bBase + kb + 256, 0, 1);
    __builtin_prefetch(aBase + kb + 256, 0, 1);

    // Preload ALL fragments so the compiler can emit one deep load clause
    // and stage s_wait_loadcnt across the 16 WMMAs (no per-group stalls).
    v16bf b[4];
#pragma unroll
    for (int ni = 0; ni < 4; ++ni)
    {
      const bf16* bp = bBase + (size_t)ni * 16 * K + kb;
      v8bf blo = *(const v8bf*)(bp);       // K: kb + half*8 .. +7
      v8bf bhi = *(const v8bf*)(bp + 16);  // K: kb + 16 + half*8 .. +7
      b[ni] = __builtin_shufflevector(blo, bhi, 0, 1, 2, 3, 4, 5, 6, 7,
                                                8, 9, 10, 11, 12, 13, 14, 15);
    }

    v16bf a[4];
#pragma unroll
    for (int mi = 0; mi < 4; ++mi) {
      const bf16* ap = aBase + (size_t)mi * 16 * K + kb;
      v8bf lo = *(const v8bf*)(ap);        // K: kb + half*8 .. +7
      v8bf hi = *(const v8bf*)(ap + 16);   // K: kb + 16 + half*8 .. +7
      a[mi] = __builtin_shufflevector(lo, hi, 0, 1, 2, 3, 4, 5, 6, 7,
                                              8, 9, 10, 11, 12, 13, 14, 15);
    }

#pragma unroll
    for (int mi = 0; mi < 4; ++mi)
#pragma unroll
      for (int ni = 0; ni < 4; ++ni)
        acc[mi][ni] = __builtin_amdgcn_wmma_f32_16x16x32_bf16(
            false, a[mi], false, b[ni], (short)0, acc[mi][ni], false, false);
  }

#pragma unroll
  for (int mi = 0; mi < 4; ++mi)
#pragma unroll
    for (int ni = 0; ni < 4; ++ni)
#pragma unroll
      for (int r = 0; r < 8; ++r) {
        const int row = tileM + mi * 16 + half * 8 + r;
        const int col = tileN + ni * 16 + lrow;
        if (STORE_BF16)
          ((bf16*)Y)[(size_t)row * N + col] = (bf16)acc[mi][ni][r];
        else
          ((float*)Y)[(size_t)row * N + col] = acc[mi][ni][r];
      }
}

// ---------------------- sliding-window char attention ----------------------
// One 64-thread block per (head h, token t). Keys j = w*CVN + c come from
// source position p = t - 7 + w (zero-padded: score 0, v 0 when p < 0).
__global__ __launch_bounds__(64) void swa_attn(const bf16* __restrict__ q,   // [T, NH*HD]
                                               const bf16* __restrict__ k,   // [T*CV, NH*HD]
                                               const bf16* __restrict__ v,   // [T*CV, NH*HD]
                                               bf16* __restrict__ out) {     // [T, NH*HD]
  const int h = blockIdx.x;
  const int t = blockIdx.y;
  const int j = threadIdx.x;          // key index (phase 1) / dim index (phase 2)

  __shared__ float sc[64];
  __shared__ float red[64];

  const int w = j >> 3;
  const int c = j & 7;
  const int p = t - (WIN - 1) + w;

  float score = 0.0f;
  if (p >= 0) {
    const v8bf* qv = (const v8bf*)(q + (size_t)t * DIM + h * HDIM);
    const v8bf* kv = (const v8bf*)(k + ((size_t)p * CVN + c) * DIM + h * HDIM);
#pragma unroll
    for (int dd = 0; dd < 8; ++dd) {
      v8bf qa = qv[dd], ka = kv[dd];
#pragma unroll
      for (int e = 0; e < 8; ++e) score += (float)qa[e] * (float)ka[e];
    }
    score *= 0.125f;  // 1/sqrt(64)
  }

  // softmax over 64 keys (padded keys keep score 0, matching the reference)
  red[j] = score;
  __syncthreads();
#pragma unroll
  for (int s = 32; s > 0; s >>= 1) {
    if (j < s) red[j] = fmaxf(red[j], red[j + s]);
    __syncthreads();
  }
  const float mx = red[0];
  __syncthreads();
  const float e = __expf(score - mx);
  sc[j] = e;
  red[j] = e;
  __syncthreads();
#pragma unroll
  for (int s = 32; s > 0; s >>= 1) {
    if (j < s) red[j] += red[j + s];
    __syncthreads();
  }
  const float inv = 1.0f / red[0];

  // phase 2: thread j = output dim d
  const int d = j;
  float o = 0.0f;
#pragma unroll 8
  for (int key = 0; key < 64; ++key) {
    const int kp = t - (WIN - 1) + (key >> 3);
    if (kp >= 0)
      o += sc[key] * (float)v[((size_t)kp * CVN + (key & 7)) * DIM + h * HDIM + d];
  }
  out[(size_t)t * DIM + h * HDIM + d] = (bf16)(o * inv);
}

// --------------------------------- launch ----------------------------------
extern "C" void kernel_launch(void* const* d_in, const int* in_sizes, int n_in,
                              void* d_out, int out_size, void* d_ws, size_t ws_size,
                              hipStream_t stream) {
  const float* x     = (const float*)d_in[0];   // [T, DIM]
  const float* chars = (const float*)d_in[1];   // [T, CV, DIM]
  const float* wq    = (const float*)d_in[2];   // [BMM, DIM]
  const float* wk    = (const float*)d_in[3];
  const float* wv    = (const float*)d_in[4];
  const float* wo    = (const float*)d_in[5];   // [DIM, BMM]
  float* out = (float*)d_out;                   // [T, DIM]

  const size_t nX  = (size_t)T_TOK * DIM;          // 2,097,152
  const size_t nC  = (size_t)T_TOK * CVN * DIM;    // 16,777,216
  const size_t nWt = (size_t)DIM * DIM;            // 4,194,304

  char* p = (char*)d_ws;
  auto take = [&](size_t elems) -> bf16* {
    bf16* r = (bf16*)p;
    p += (elems * sizeof(bf16) + 255) & ~(size_t)255;
    return r;
  };
  bf16* xb  = take(nX);
  bf16* cb  = take(nC);
  bf16* wqb = take(nWt);
  bf16* wkb = take(nWt);
  bf16* wvb = take(nWt);
  bf16* wob = take(nWt);
  bf16* qb  = take(nX);   // q  [T, NH*HD]
  bf16* kb  = take(nC);   // k  [T*CV, NH*HD]
  bf16* vb  = take(nC);   // v  [T*CV, NH*HD]
  bf16* ab  = take(nX);   // attention output [T, NH*HD]

  auto cvt = [&](const float* s, bf16* d, size_t n) {
    int n4 = (int)(n / 4);
    cvt_f32_bf16<<<(n4 + 255) / 256, 256, 0, stream>>>(s, d, n4);
  };
  cvt(x, xb, nX);
  cvt(chars, cb, nC);
  cvt(wq, wqb, nWt);
  cvt(wk, wkb, nWt);
  cvt(wv, wvb, nWt);
  cvt(wo, wob, nWt);

  // Projections: Y[M,2048] = X[M,2048] @ W[2048,2048]^T  (bf16 out)
  dim3 blk(256);
  gemm_bf16<1><<<dim3(T_TOK / 128, DIM / 256), blk, 0, stream>>>(xb, wqb, qb, T_TOK, DIM, DIM);
  gemm_bf16<1><<<dim3(T_TOK * CVN / 128, DIM / 256), blk, 0, stream>>>(cb, wkb, kb, T_TOK * CVN, DIM, DIM);
  gemm_bf16<1><<<dim3(T_TOK * CVN / 128, DIM / 256), blk, 0, stream>>>(cb, wvb, vb, T_TOK * CVN, DIM, DIM);

  // Sliding-window attention (RoPE cancels exactly; skipped)
  swa_attn<<<dim3(NHEAD, T_TOK), 64, 0, stream>>>(qb, kb, vb, ab);

  // Output projection (f32 out -> d_out)
  gemm_bf16<0><<<dim3(T_TOK / 128, DIM / 256), blk, 0, stream>>>(ab, wob, out, T_TOK, DIM, DIM);
}
